// CLIP4CAD_HUS_v2_64287070486951
// MI455X (gfx1250) — compile-verified, matrix operand order checked by reading, not run on hardware
//
#include <hip/hip_runtime.h>
#include <hip/hip_bf16.h>

#define D_   512
#define H_   8
#define HD_  64
#define G_   4
#define M_   16
#define DF_  2048
#define B_   64
#define N_   2048
#define TN   16                 // token tile
#define XSTR (D_ + 16)          // bf16 X tile stride
#define KVSTR (HD_ + 8)
#define PSTR 40                 // 32 (padded K) + 8

typedef __attribute__((ext_vector_type(16))) __bf16 v16bf;
typedef __attribute__((ext_vector_type(8)))  float  v8f;

// -------- CDNA5 async global->LDS path (probe-confirmed in round 3) -------
#if defined(__has_builtin)
#  if __has_builtin(__builtin_amdgcn_global_load_async_to_lds_b128)
#    define ATHENA_ASYNC 1
#  endif
#endif
#ifndef ATHENA_ASYNC
#  define ATHENA_ASYNC 0
#endif

#if ATHENA_ASYNC
// signature (from compiler diagnostic): (v4i addrspace(1)*, v4i addrspace(3)*, Iimm, Iimm)
typedef int v4i_gcc __attribute__((vector_size(4 * sizeof(int))));
typedef __attribute__((address_space(1))) v4i_gcc* gptr_v4i;
typedef __attribute__((address_space(3))) v4i_gcc* lptr_v4i;
__device__ __forceinline__ void async_ld_b128(const float* g, float* l) {
  __builtin_amdgcn_global_load_async_to_lds_b128(
      (gptr_v4i)(unsigned long long)g,
      (lptr_v4i)(unsigned int)(unsigned long long)l, 0, 0);
}
__device__ __forceinline__ void async_wait0() {
#  if __has_builtin(__builtin_amdgcn_s_wait_asynccnt)
  __builtin_amdgcn_s_wait_asynccnt(0);
#  else
  asm volatile("s_wait_asynccnt 0" ::: "memory");
#  endif
}
#endif

__device__ __forceinline__ int kk_of(int i, int hi) {
  return ((i < 4) ? (2 * i) : (16 + 2 * (i - 4))) + hi * 8;
}

// A-fragment from a 16-row row-major bf16 source; also the B-fragment loader
// for B = W^T with W row-major (out_dim, in_dim).
__device__ __forceinline__ v16bf frag_rows(const __bf16* p, int stride, int rowbase, int kbase) {
  int lane = threadIdx.x & 31;
  int hi = lane >> 4;
  const __bf16* r = p + (size_t)(rowbase + (lane & 15)) * stride + kbase;
  v16bf f;
#pragma unroll
  for (int i = 0; i < 8; ++i) {
    int kk = kk_of(i, hi);
    f[2 * i]     = r[kk];
    f[2 * i + 1] = r[kk + 1];
  }
  return f;
}

// B-fragment from a K-major (K x N, row-major) bf16 source.
__device__ __forceinline__ v16bf frag_kmajor(const __bf16* p, int stride, int colbase, int kbase) {
  int lane = threadIdx.x & 31;
  int hi = lane >> 4;
  const __bf16* c = p + colbase + (lane & 15);
  v16bf f;
#pragma unroll
  for (int i = 0; i < 8; ++i) {
    int kk = kk_of(i, hi) + kbase;
    f[2 * i]     = c[(size_t)kk * stride];
    f[2 * i + 1] = c[(size_t)(kk + 1) * stride];
  }
  return f;
}

__device__ __forceinline__ v8f wmma_bf16(v16bf a, v16bf b, v8f c) {
  return __builtin_amdgcn_wmma_f32_16x16x32_bf16(false, a, false, b, (short)0, c, false, false);
}

__device__ __forceinline__ float gelu_f(float x) {
  return 0.5f * x * (1.0f + erff(x * 0.70710678118654752f));
}

// ---------------------------------------------------------------------------
// Kernel 1: small prep — adapt, adapted queries, q_global (bf16, scaled,
// zero-padded to 16 rows), qd_base (f32, unscaled, modn added later).
// ---------------------------------------------------------------------------
__global__ __launch_bounds__(512) void prep_k(
    const float* __restrict__ mod_embed, const float* __restrict__ adapt_w,
    const float* __restrict__ adapt_b,  const float* __restrict__ gq,
    const float* __restrict__ dq,       const float* __restrict__ mha_in_w,
    const float* __restrict__ mha_in_b, const float* __restrict__ det_wq,
    const float* __restrict__ det_bq,
    float* __restrict__ gq_eff, float* __restrict__ qd_base,
    __bf16* __restrict__ qg)
{
  __shared__ float sadapt[D_];
  __shared__ float sgq[G_ * D_];
  __shared__ float sdq[M_ * D_];
  int t = threadIdx.x;
  {
    const float* me = mod_embed + D_;               // modality index 1
    const float* wr = adapt_w + (size_t)t * D_;
    float acc = adapt_b[t];
    for (int d = 0; d < D_; ++d) acc += me[d] * wr[d];
    sadapt[t] = tanhf(acc);
  }
  __syncthreads();
  for (int i = t; i < G_ * D_; i += 512) {
    float v = gq[i] + 0.1f * sadapt[i & (D_ - 1)];
    sgq[i] = v; gq_eff[i] = v;
  }
  for (int i = t; i < M_ * D_; i += 512)
    sdq[i] = dq[i] + 0.1f * sadapt[i & (D_ - 1)];
  for (int i = t; i < H_ * 16 * HD_; i += 512) qg[i] = (__bf16)0.0f;
  __syncthreads();
  for (int i = t; i < G_ * D_; i += 512) {         // q = gq_eff @ wq^T + bq
    int g = i >> 9, j = i & (D_ - 1);
    const float* wr = mha_in_w + (size_t)j * D_;   // wq rows 0..511
    const float* zr = sgq + g * D_;
    float acc = mha_in_b[j];
    for (int d = 0; d < D_; ++d) acc += zr[d] * wr[d];
    int h = j >> 6, hd = j & 63;
    qg[h * (16 * HD_) + g * HD_ + hd] = (__bf16)(0.125f * acc);  // fold 1/sqrt(64)
  }
  for (int i = t; i < M_ * D_; i += 512) {         // qd_base = dq_eff @ det_wq^T + det_bq
    int m = i >> 9, j = i & (D_ - 1);
    const float* wr = det_wq + (size_t)j * D_;
    const float* zr = sdq + m * D_;
    float acc = det_bq[j];
    for (int d = 0; d < D_; ++d) acc += zr[d] * wr[d];
    qd_base[i] = acc;
  }
}

// ---------------------------------------------------------------------------
// Kernel 2/4: fused projection + flash attention, one block per (head, batch).
// Streams 16-token tiles of X; tile staged by async DMA into LDS (f32),
// converted once per wave to bf16; K/V projection, scores and O accumulation
// on v_wmma_f32_16x16x32_bf16. DMA of tile t+1 overlaps phases B..E of tile t.
// ---------------------------------------------------------------------------
__global__ __launch_bounds__(256) void attn_k(
    const float* __restrict__ X, const unsigned char* __restrict__ mask,
    const float* __restrict__ Wk, const float* __restrict__ bk,
    const float* __restrict__ Wv, const float* __restrict__ bv,
    const __bf16* __restrict__ Q, int q_batch_stride,
    float* __restrict__ o_acc, float* __restrict__ o_sum)
{
  int h = blockIdx.x, b = blockIdx.y;
  int tid = threadIdx.x, wave = tid >> 5, lane = tid & 31;

  __shared__ float  lxf[TN * D_];           // DMA landing pad (f32)
  __shared__ __bf16 lx[TN * XSTR];          // X tile (bf16)
  __shared__ __bf16 lk[TN * KVSTR];         // K tile (tok, hd)
  __shared__ __bf16 lv[32 * KVSTR];         // V tile, rows 16..31 zero pad
  __shared__ __bf16 lq[16 * HD_];           // queries (scaled)
  __shared__ __bf16 lp[16 * PSTR];          // probs, cols 16..31 zero pad
  __shared__ float  ls[16 * TN];            // raw scores
  __shared__ float  lmax[16], lsum[16], lscale[16];
  __shared__ unsigned char lmask[TN];

  const __bf16* qsrc = Q + (size_t)b * q_batch_stride + (size_t)h * (16 * HD_);
  for (int i = tid; i < 16 * HD_; i += 256) lq[i] = qsrc[i];
  for (int i = tid; i < 16 * 16; i += 256)                     // zero P pad cols
    lp[(i >> 4) * PSTR + 16 + (i & 15)] = (__bf16)0.0f;
  for (int i = tid; i < 16 * HD_; i += 256)                    // zero V pad rows
    lv[(16 + (i >> 6)) * KVSTR + (i & 63)] = (__bf16)0.0f;
  if (tid < 16) { lmax[tid] = -3.0e38f; lsum[tid] = 0.0f; }

  // each wave owns one (projection, 16-col slice); weight B-fragments in VGPRs
  int proj = wave >> 2;                     // 0 -> K, 1 -> V
  int j16  = wave & 3;
  int ncol = j16 * 16 + (lane & 15);
  const float* W = proj ? Wv : Wk;
  float biasn = (proj ? bv : bk)[h * HD_ + ncol];
  v16bf wfrag[16];
  {
    int hi = lane >> 4;
    const float* wr = W + (size_t)(h * HD_ + ncol) * D_;
#pragma unroll
    for (int c = 0; c < 16; ++c) {
      v16bf f;
#pragma unroll
      for (int i = 0; i < 8; ++i) {
        int kk = c * 32 + kk_of(i, hi);
        f[2 * i]     = (__bf16)wr[kk];
        f[2 * i + 1] = (__bf16)wr[kk + 1];
      }
      wfrag[c] = f;
    }
  }
  __syncthreads();
  v16bf qfrag0 = frag_rows(lq, HD_, 0, 0);
  v16bf qfrag1 = frag_rows(lq, HD_, 0, 32);
  v8f oaccv = {0.f,0.f,0.f,0.f,0.f,0.f,0.f,0.f};

  const float* Xb = X + (size_t)b * N_ * D_;
  const unsigned char* mb = mask + (size_t)b * N_;

  // stage(tile): each wave fetches its own 2 rows (async DMA or prefetch)
  auto stage = [&](int tnext) {
#if ATHENA_ASYNC
#pragma unroll
    for (int q = 0; q < 8; ++q) {
      int r  = 2 * wave + (q >> 2);
      int cc = (q & 3) * 32 + lane;          // float4 index within row
      async_ld_b128(Xb + (size_t)(tnext + r) * D_ + cc * 4, lxf + r * D_ + cc * 4);
    }
#else
    __builtin_prefetch(Xb + (size_t)tnext * D_ + tid * 32, 0, 0);
#endif
  };

  stage(0);
  for (int t0 = 0; t0 < N_; t0 += TN) {
    // phase A: convert own rows f32 -> bf16 (async: from LDS landing pad)
#if ATHENA_ASYNC
    async_wait0();
#pragma unroll
    for (int q = 0; q < 8; ++q) {
      int r  = 2 * wave + (q >> 2);
      int cc = (q & 3) * 32 + lane;
      const float4 xv = ((const float4*)(lxf + r * D_))[cc];
      __bf16* o = lx + r * XSTR + cc * 4;
      o[0] = (__bf16)xv.x; o[1] = (__bf16)xv.y;
      o[2] = (__bf16)xv.z; o[3] = (__bf16)xv.w;
    }
#else
#pragma unroll
    for (int q = 0; q < 8; ++q) {
      int r  = 2 * wave + (q >> 2);
      int cc = (q & 3) * 32 + lane;
      const float4 xv = ((const float4*)(Xb + (size_t)(t0 + r) * D_))[cc];
      __bf16* o = lx + r * XSTR + cc * 4;
      o[0] = (__bf16)xv.x; o[1] = (__bf16)xv.y;
      o[2] = (__bf16)xv.z; o[3] = (__bf16)xv.w;
    }
#endif
    if (tid < TN) lmask[tid] = mb[t0 + tid];
    if (t0 + TN < N_) stage(t0 + TN);        // DMA overlaps phases B..E
    __syncthreads();

    // phase B: K/V projection; two independent accumulator chains so the
    // scheduler can overlap DS-load latency with the matrix pipe
    {
      v8f c0 = {0.f,0.f,0.f,0.f,0.f,0.f,0.f,0.f};
      v8f c1 = {0.f,0.f,0.f,0.f,0.f,0.f,0.f,0.f};
      v16bf a0 = frag_rows(lx, XSTR, 0, 0);
      v16bf a1 = frag_rows(lx, XSTR, 0, 32);
#pragma unroll
      for (int kc = 0; kc < 16; kc += 2) {
        v16bf a2 = frag_rows(lx, XSTR, 0, ((kc + 2) & 15) * 32);
        c0 = wmma_bf16(a0, wfrag[kc], c0);
        v16bf a3 = frag_rows(lx, XSTR, 0, ((kc + 3) & 15) * 32);
        c1 = wmma_bf16(a1, wfrag[kc + 1], c1);
        a0 = a2; a1 = a3;
      }
      __bf16* dst = proj ? lv : lk;
      int hi = lane >> 4;
#pragma unroll
      for (int i = 0; i < 8; ++i)
        dst[(i + hi * 8) * KVSTR + ncol] = (__bf16)(c0[i] + c1[i] + biasn);
    }
    __syncthreads();

    // phase C: S = Q @ K^T (scale folded into Q), masked, to LDS f32
    if (wave == 0) {
      v8f s = {0.f,0.f,0.f,0.f,0.f,0.f,0.f,0.f};
      v16bf k0 = frag_rows(lk, KVSTR, 0, 0);
      v16bf k1 = frag_rows(lk, KVSTR, 0, 32);
      s = wmma_bf16(qfrag0, k0, s);
      s = wmma_bf16(qfrag1, k1, s);
      int tok = lane & 15;
      bool keep = lmask[tok] != 0;
      int hi = lane >> 4;
#pragma unroll
      for (int i = 0; i < 8; ++i)
        ls[(i + hi * 8) * TN + tok] = keep ? s[i] : -1.0e30f;
    }
    __syncthreads();

    // phase D: online softmax update (16 rows)
    if (tid < 16) {
      float mo = lmax[tid], mx = mo;
      const float* row = ls + tid * TN;
      for (int n = 0; n < TN; ++n) mx = fmaxf(mx, row[n]);
      float corr = __expf(mo - mx);
      float sum = lsum[tid] * corr;
      __bf16* pr = lp + tid * PSTR;
      for (int n = 0; n < TN; ++n) {
        float p = __expf(row[n] - mx);
        sum += p;
        pr[n] = (__bf16)p;
      }
      lmax[tid] = mx; lsum[tid] = sum; lscale[tid] = corr;
    }
    __syncthreads();

    // phase E: O = O*scale + P @ V (K padded to 32 with zero P cols / V rows)
    if (wave >= 4) {
      int hi = lane >> 4;
#pragma unroll
      for (int i = 0; i < 8; ++i) oaccv[i] *= lscale[i + hi * 8];
      oaccv = wmma_bf16(frag_rows(lp, PSTR, 0, 0),
                        frag_kmajor(lv, KVSTR, (wave - 4) * 16, 0), oaccv);
    }
  }
  __syncthreads();
  if (wave >= 4) {
    int hi = lane >> 4;
    float* dst = o_acc + ((size_t)(b * H_ + h) * 16) * HD_;
#pragma unroll
    for (int i = 0; i < 8; ++i)
      dst[(i + hi * 8) * HD_ + (wave - 4) * 16 + (lane & 15)] = oaccv[i];
  }
  if (tid < 16) o_sum[(size_t)(b * H_ + h) * 16 + tid] = lsum[tid];
}

// ---------------------------------------------------------------------------
// Kernel 3: finish global branch, produce z_global/zgp/cond/modn/qd (bf16).
// ---------------------------------------------------------------------------
__global__ __launch_bounds__(1024) void mid_k(
    const float* __restrict__ o_acc, const float* __restrict__ o_sum,
    const float* __restrict__ gq_eff,
    const float* __restrict__ Wo, const float* __restrict__ bo,
    const float* __restrict__ g1g, const float* __restrict__ g1b,
    const float* __restrict__ W1, const float* __restrict__ b1,
    const float* __restrict__ W2, const float* __restrict__ b2,
    const float* __restrict__ g2g, const float* __restrict__ g2b,
    const float* __restrict__ cw1, const float* __restrict__ cb1,
    const float* __restrict__ cw2, const float* __restrict__ cb2,
    const float* __restrict__ qd_base, __bf16* __restrict__ qd,
    float* __restrict__ zgp_ws, float* __restrict__ out)
{
  int b = blockIdx.x, t = threadIdx.x;
  __shared__ float so[G_ * D_];
  __shared__ float zg[G_ * D_];
  __shared__ float h1[G_ * DF_];
  __shared__ float scond[D_], sc1[D_];
  __shared__ float smu[G_], srs[G_];

  for (int i = t; i < G_ * D_; i += 1024) {        // normalized attention out
    int g = i >> 9, d = i & (D_ - 1);
    int h = d >> 6, hd = d & 63;
    size_t base = (size_t)(b * H_ + h) * 16 + g;
    so[i] = o_acc[base * HD_ + hd] / o_sum[base];
  }
  __syncthreads();
  for (int i = t; i < G_ * D_; i += 1024) {        // out-proj + residual gq
    int g = i >> 9, j = i & (D_ - 1);
    const float* wr = Wo + (size_t)j * D_;
    const float* xr = so + g * D_;
    float acc = bo[j];
    for (int d = 0; d < D_; ++d) acc += xr[d] * wr[d];
    zg[i] = acc + gq_eff[i];
  }
  __syncthreads();
  if (t < G_) {
    float mu = 0.f, v = 0.f;
    const float* r = zg + t * D_;
    for (int d = 0; d < D_; ++d) mu += r[d];
    mu /= (float)D_;
    for (int d = 0; d < D_; ++d) { float x = r[d] - mu; v += x * x; }
    smu[t] = mu; srs[t] = rsqrtf(v / (float)D_ + 1e-5f);
  }
  __syncthreads();
  for (int i = t; i < G_ * D_; i += 1024) {        // LN1
    int g = i >> 9, j = i & (D_ - 1);
    zg[i] = (zg[i] - smu[g]) * srs[g] * g1g[j] + g1b[j];
  }
  __syncthreads();
  for (int i = t; i < G_ * DF_; i += 1024) {       // FFN up + gelu
    int g = i >> 11, f = i & (DF_ - 1);
    const float* wr = W1 + (size_t)f * D_;
    const float* xr = zg + g * D_;
    float acc = b1[f];
    for (int d = 0; d < D_; ++d) acc += xr[d] * wr[d];
    h1[i] = gelu_f(acc);
  }
  __syncthreads();
  for (int i = t; i < G_ * D_; i += 1024) {        // FFN down + residual
    int g = i >> 9, j = i & (D_ - 1);
    const float* wr = W2 + (size_t)j * DF_;
    const float* xr = h1 + g * DF_;
    float acc = b2[j];
    for (int f = 0; f < DF_; ++f) acc += xr[f] * wr[f];
    so[i] = acc + zg[i];
  }
  __syncthreads();
  if (t < G_) {
    float mu = 0.f, v = 0.f;
    const float* r = so + t * D_;
    for (int d = 0; d < D_; ++d) mu += r[d];
    mu /= (float)D_;
    for (int d = 0; d < D_; ++d) { float x = r[d] - mu; v += x * x; }
    smu[t] = mu; srs[t] = rsqrtf(v / (float)D_ + 1e-5f);
  }
  __syncthreads();
  for (int i = t; i < G_ * D_; i += 1024) {        // LN2 -> z_global
    int g = i >> 9, j = i & (D_ - 1);
    so[i] = (so[i] - smu[g]) * srs[g] * g2g[j] + g2b[j];
  }
  __syncthreads();
  if (t < D_) {                                    // zgp == cond
    float s = 0.25f * (so[t] + so[D_ + t] + so[2 * D_ + t] + so[3 * D_ + t]);
    scond[t] = s;
    zgp_ws[(size_t)b * D_ + t] = s;
    out[(size_t)b * D_ + t] = s;                   // output row 0
  }
  __syncthreads();
  if (t < D_) {                                    // cond MLP layer 1
    const float* wr = cw1 + (size_t)t * D_;
    float acc = cb1[t];
    for (int d = 0; d < D_; ++d) acc += scond[d] * wr[d];
    sc1[t] = gelu_f(acc);
  }
  __syncthreads();
  if (t < D_) {                                    // modn (reuse zg row 0)
    const float* wr = cw2 + (size_t)t * D_;
    float acc = cb2[t];
    for (int d = 0; d < D_; ++d) acc += sc1[d] * wr[d];
    zg[t] = acc;
  }
  __syncthreads();
  for (int i = t; i < M_ * D_; i += 1024) {        // scaled bf16 detail queries
    int m = i >> 9, j = i & (D_ - 1);
    int h = j >> 6, hd = j & 63;
    qd[((size_t)(b * H_ + h) * 16 + m) * HD_ + hd] =
        (__bf16)(0.125f * (qd_base[i] + zg[j]));
  }
}

// ---------------------------------------------------------------------------
// Kernel 5: finish detail branch + gated fusion, output rows 1 and 2.
// ---------------------------------------------------------------------------
__global__ __launch_bounds__(1024) void fuse_k(
    const float* __restrict__ od_acc, const float* __restrict__ od_sum,
    const float* __restrict__ Wo, const float* __restrict__ bo,
    const float* __restrict__ W1, const float* __restrict__ b1,
    const float* __restrict__ W2, const float* __restrict__ b2,
    const float* __restrict__ dng, const float* __restrict__ dnb,
    const float* __restrict__ zgp_ws,
    const float* __restrict__ gw1, const float* __restrict__ gb1,
    const float* __restrict__ gw2, const float* __restrict__ gb2,
    const float* __restrict__ fw, const float* __restrict__ fb,
    const float* __restrict__ fg, const float* __restrict__ fbb,
    float* __restrict__ out)
{
  int b = blockIdx.x, t = threadIdx.x;
  __shared__ float bufA[M_ * 256];
  __shared__ float bufB[M_ * D_];
  __shared__ float szgp[D_], szdp[D_], sg1[D_], sfus[D_];
  __shared__ float smu[M_], srs[M_];
  __shared__ float sgate[2], sfstat[2];

  float acc8[8];
#pragma unroll
  for (int r = 0; r < 8; ++r) acc8[r] = 0.0f;

  // z_detail_pre = od @ det_wo^T + bo  (od streamed in 128-wide d chunks)
  for (int dc = 0; dc < 4; ++dc) {
    __syncthreads();
    for (int i = t; i < M_ * 128; i += 1024) {
      int m = i >> 7, dl = i & 127;
      int d = dc * 128 + dl;
      int h = d >> 6, hd = d & 63;
      size_t base = (size_t)(b * H_ + h) * 16 + m;
      bufA[m * 128 + dl] = od_acc[base * HD_ + hd] / od_sum[base];
    }
    __syncthreads();
#pragma unroll
    for (int r = 0; r < 8; ++r) {
      int idx = t + r * 1024;
      int m = idx >> 9, j = idx & (D_ - 1);
      const float* wr = Wo + (size_t)j * D_ + dc * 128;
      const float* xr = bufA + m * 128;
      float a = acc8[r];
      for (int dl = 0; dl < 128; ++dl) a += xr[dl] * wr[dl];
      acc8[r] = a;
    }
  }
  __syncthreads();
#pragma unroll
  for (int r = 0; r < 8; ++r) {
    int idx = t + r * 1024;
    bufB[idx] = acc8[r] + bo[idx & (D_ - 1)];
  }
  // detail FFN, 256-wide f chunks; y2 accumulates in registers
  float y2[8];
#pragma unroll
  for (int r = 0; r < 8; ++r) y2[r] = 0.0f;
  for (int fc = 0; fc < 8; ++fc) {
    __syncthreads();
    for (int i = t; i < M_ * 256; i += 1024) {
      int m = i >> 8, fl = i & 255;
      int f = fc * 256 + fl;
      const float* wr = W1 + (size_t)f * D_;
      const float* xr = bufB + m * D_;
      float acc = b1[f];
      for (int d = 0; d < D_; ++d) acc += xr[d] * wr[d];
      bufA[i] = gelu_f(acc);
    }
    __syncthreads();
#pragma unroll
    for (int r = 0; r < 8; ++r) {
      int idx = t + r * 1024;
      int m = idx >> 9, j = idx & (D_ - 1);
      const float* wr = W2 + (size_t)j * DF_ + fc * 256;
      const float* xr = bufA + m * 256;
      float a = y2[r];
      for (int fl = 0; fl < 256; ++fl) a += xr[fl] * wr[fl];
      y2[r] = a;
    }
  }
  __syncthreads();
#pragma unroll
  for (int r = 0; r < 8; ++r) {
    int idx = t + r * 1024;
    bufB[idx] += y2[r] + b2[idx & (D_ - 1)];
  }
  __syncthreads();
  if (t < M_) {
    float mu = 0.f, v = 0.f;
    const float* r = bufB + t * D_;
    for (int d = 0; d < D_; ++d) mu += r[d];
    mu /= (float)D_;
    for (int d = 0; d < D_; ++d) { float x = r[d] - mu; v += x * x; }
    smu[t] = mu; srs[t] = rsqrtf(v / (float)D_ + 1e-5f);
  }
  __syncthreads();
#pragma unroll
  for (int r = 0; r < 8; ++r) {
    int idx = t + r * 1024;
    int m = idx >> 9, j = idx & (D_ - 1);
    bufB[idx] = (bufB[idx] - smu[m]) * srs[m] * dng[j] + dnb[j];
  }
  __syncthreads();
  if (t < D_) {
    float s = 0.0f;
    for (int m = 0; m < M_; ++m) s += bufB[m * D_ + t];
    s *= (1.0f / 16.0f);
    szdp[t] = s;
    out[(size_t)(B_ * D_) + (size_t)b * D_ + t] = s;     // output row 1
    szgp[t] = zgp_ws[(size_t)b * D_ + t];
  }
  __syncthreads();
  if (t < D_) {
    const float* wr = gw1 + (size_t)t * (2 * D_);
    float acc = gb1[t];
    for (int d = 0; d < D_; ++d) acc += szgp[d] * wr[d] + szdp[d] * wr[D_ + d];
    sg1[t] = gelu_f(acc);
    const float* fr = fw + (size_t)t * (2 * D_);
    float fa = fb[t];
    for (int d = 0; d < D_; ++d) fa += szgp[d] * fr[d] + szdp[d] * fr[D_ + d];
    sfus[t] = gelu_f(fa);
  }
  __syncthreads();
  if (t == 0) {
    float l0 = gb2[0], l1 = gb2[1];
    for (int d = 0; d < D_; ++d) { l0 += sg1[d] * gw2[d]; l1 += sg1[d] * gw2[D_ + d]; }
    float mx = fmaxf(l0, l1);
    float e0 = __expf(l0 - mx), e1 = __expf(l1 - mx);
    float inv = 1.0f / (e0 + e1);
    sgate[0] = e0 * inv; sgate[1] = e1 * inv;
    float mu = 0.f;
    for (int d = 0; d < D_; ++d) mu += sfus[d];
    mu /= (float)D_;
    float v = 0.f;
    for (int d = 0; d < D_; ++d) { float x = sfus[d] - mu; v += x * x; }
    sfstat[0] = mu; sfstat[1] = rsqrtf(v / (float)D_ + 1e-5f);
  }
  __syncthreads();
  if (t < D_) {
    float zu = (sfus[t] - sfstat[0]) * sfstat[1] * fg[t] + fbb[t]
             + sgate[0] * szgp[t] + sgate[1] * szdp[t];
    out[2 * (size_t)(B_ * D_) + (size_t)b * D_ + t] = zu; // output row 2
  }
}

// ---------------------------------------------------------------------------
extern "C" void kernel_launch(void* const* d_in, const int* in_sizes, int n_in,
                              void* d_out, int out_size, void* d_ws, size_t ws_size,
                              hipStream_t stream) {
  (void)in_sizes; (void)n_in; (void)out_size; (void)ws_size;
  const float* X             = (const float*)d_in[0];
  const unsigned char* mask  = (const unsigned char*)d_in[1];
  const float* gq            = (const float*)d_in[2];
  const float* dq            = (const float*)d_in[3];
  const float* mod_embed     = (const float*)d_in[4];
  const float* adapt_w       = (const float*)d_in[5];
  const float* adapt_b       = (const float*)d_in[6];
  const float* mha_in_w      = (const float*)d_in[7];
  const float* mha_in_b      = (const float*)d_in[8];
  const float* mha_out_w     = (const float*)d_in[9];
  const float* mha_out_b     = (const float*)d_in[10];
  const float* gffn_w1       = (const float*)d_in[11];
  const float* gffn_b1       = (const float*)d_in[12];
  const float* gffn_w2       = (const float*)d_in[13];
  const float* gffn_b2       = (const float*)d_in[14];
  const float* gn1_g         = (const float*)d_in[15];
  const float* gn1_b         = (const float*)d_in[16];
  const float* gn2_g         = (const float*)d_in[17];
  const float* gn2_b         = (const float*)d_in[18];
  const float* det_wq        = (const float*)d_in[19];
  const float* det_bq        = (const float*)d_in[20];
  const float* det_wk        = (const float*)d_in[21];
  const float* det_bk        = (const float*)d_in[22];
  const float* det_wv        = (const float*)d_in[23];
  const float* det_bv        = (const float*)d_in[24];
  const float* det_wo        = (const float*)d_in[25];
  const float* det_bo        = (const float*)d_in[26];
  const float* cond_w1       = (const float*)d_in[27];
  const float* cond_b1       = (const float*)d_in[28];
  const float* cond_w2       = (const float*)d_in[29];
  const float* cond_b2       = (const float*)d_in[30];
  const float* dffn_w1       = (const float*)d_in[31];
  const float* dffn_b1       = (const float*)d_in[32];
  const float* dffn_w2       = (const float*)d_in[33];
  const float* dffn_b2       = (const float*)d_in[34];
  const float* dn_g          = (const float*)d_in[35];
  const float* dn_b          = (const float*)d_in[36];
  const float* gate_w1       = (const float*)d_in[37];
  const float* gate_b1       = (const float*)d_in[38];
  const float* gate_w2       = (const float*)d_in[39];
  const float* gate_b2       = (const float*)d_in[40];
  const float* fus_w         = (const float*)d_in[41];
  const float* fus_b         = (const float*)d_in[42];
  const float* fus_g         = (const float*)d_in[43];
  const float* fus_bb        = (const float*)d_in[44];
  float* out = (float*)d_out;

  char* ws = (char*)d_ws;
  size_t off = 0;
  auto take = [&](size_t bytes) {
    char* p = ws + off;
    off += (bytes + 255) & ~(size_t)255;
    return p;
  };
  __bf16* qg     = (__bf16*)take((size_t)H_ * 16 * HD_ * sizeof(__bf16));
  __bf16* qd     = (__bf16*)take((size_t)B_ * H_ * 16 * HD_ * sizeof(__bf16));
  float* gq_eff  = (float*)take((size_t)G_ * D_ * sizeof(float));
  float* qd_base = (float*)take((size_t)M_ * D_ * sizeof(float));
  float* o_acc   = (float*)take((size_t)B_ * H_ * 16 * HD_ * sizeof(float));
  float* o_sum   = (float*)take((size_t)B_ * H_ * 16 * sizeof(float));
  float* od_acc  = (float*)take((size_t)B_ * H_ * 16 * HD_ * sizeof(float));
  float* od_sum  = (float*)take((size_t)B_ * H_ * 16 * sizeof(float));
  float* zgp_ws  = (float*)take((size_t)B_ * D_ * sizeof(float));

  prep_k<<<1, 512, 0, stream>>>(mod_embed, adapt_w, adapt_b, gq, dq,
                                mha_in_w, mha_in_b, det_wq, det_bq,
                                gq_eff, qd_base, qg);
  attn_k<<<dim3(H_, B_), 256, 0, stream>>>(X, mask,
      mha_in_w + (size_t)D_ * D_, mha_in_b + D_,
      mha_in_w + 2 * (size_t)D_ * D_, mha_in_b + 2 * D_,
      qg, 0, o_acc, o_sum);
  mid_k<<<B_, 1024, 0, stream>>>(o_acc, o_sum, gq_eff, mha_out_w, mha_out_b,
      gn1_g, gn1_b, gffn_w1, gffn_b1, gffn_w2, gffn_b2, gn2_g, gn2_b,
      cond_w1, cond_b1, cond_w2, cond_b2, qd_base, qd, zgp_ws, out);
  attn_k<<<dim3(H_, B_), 256, 0, stream>>>(X, mask, det_wk, det_bk,
      det_wv, det_bv, qd, H_ * 16 * HD_, od_acc, od_sum);
  fuse_k<<<B_, 1024, 0, stream>>>(od_acc, od_sum, det_wo, det_bo,
      dffn_w1, dffn_b1, dffn_w2, dffn_b2, dn_g, dn_b, zgp_ws,
      gate_w1, gate_b1, gate_w2, gate_b2, fus_w, fus_b, fus_g, fus_bb, out);
}
